// MACE_65618510348697
// MI455X (gfx1250) — matrix operand order repaired and checked by reading, not compiled, not run on hardware
//
#include <hip/hip_runtime.h>

typedef __attribute__((ext_vector_type(16))) _Float16 v16h;
typedef __attribute__((ext_vector_type(8)))  float    v8f;

#define KN      10000
#define KE      160000
#define KG      16
#define KF      32
#define AVG_INV 0.025f
#define INV_CUT (1.0f/6.0f)

// packed-weight offsets (in halves) within one layer's pack region
#define OFF_W1F   0        // K=32(pad from 8) x N=64  : 1x4 frags
#define OFF_W2F   2048     // K=64 x N=64              : 2x4
#define OFF_W3F   6144     // K=64 x N=64              : 2x4
#define OFF_W4F   10240    // K=64 x N=128             : 2x8
#define OFF_W4T   18432    // K=128 x N=64             : 4x4
#define OFF_W3T   26624    // K=64 x N=64              : 2x4
#define OFF_W2T   30720    // K=64 x N=64              : 2x4
#define OFF_W1T   34816    // K=64 x N=16(pad from 8)  : 2x1
#define OFF_WMIX  35840    // 4 x (K=32 x N=32 : 1x2)
#define OFF_WMIXT 39936    // 4 x (K=32 x N=32 : 1x2)
#define PACK_PER_LAYER 44032

__device__ inline v8f v8zero() { v8f c = {0.f,0.f,0.f,0.f,0.f,0.f,0.f,0.f}; return c; }

__device__ inline v8f wmma16(v16h a, v16h b, v8f c) {
  return __builtin_amdgcn_wmma_f32_16x16x32_f16(false, a, false, b, (short)0, c, false, false);
}

// B fragment pre-packed as [frag][lane][16 halves] -> one coalesced 32B load
__device__ inline v16h bfrag(const _Float16* base, int frag, int lane) {
  return *(const v16h*)(base + (size_t)((frag << 5) + lane) * 16);
}

// A fragment (16x32 slice at kt) from a row-major [16][width] f16 LDS tile
__device__ inline v16h afrag_h(const _Float16* t, int width, int kt, int lane) {
  int row = lane & 15, hi = (lane >> 4) & 1;
  v16h a;
#pragma unroll
  for (int e = 0; e < 16; ++e) {
    int K = kt * 32 + e + ((e & 8) ? 8 : 0) + hi * 8;
    a[e] = t[row * width + K];
  }
  return a;
}

__device__ inline v16h afrag_f(const float* t, int width, int kt, int lane) {
  int row = lane & 15, hi = (lane >> 4) & 1;
  v16h a;
#pragma unroll
  for (int e = 0; e < 16; ++e) {
    int K = kt * 32 + e + ((e & 8) ? 8 : 0) + hi * 8;
    a[e] = (_Float16)t[row * width + K];
  }
  return a;
}

__device__ inline float siluf(float x)  { float s = 1.f/(1.f+__expf(-x)); return x*s; }
__device__ inline float dsiluf(float x) { float s = 1.f/(1.f+__expf(-x)); return s*(1.f + x*(1.f-s)); }

// ---------------------------------------------------------------- weight pack
__device__ inline void packB(const float* src, int rows, int cols, bool transpose,
                             int KT, int NT, _Float16* dst, int tid, int nth) {
  int total = KT * NT * 512;
  for (int idx = tid; idx < total; idx += nth) {
    int e = idx & 15, lane = (idx >> 4) & 31, frag = idx >> 9;
    int nt = frag % NT;
    int kt = frag / NT;
    int K = kt * 32 + e + ((e & 8) ? 8 : 0) + ((lane & 16) ? 8 : 0);
    int n = nt * 16 + (lane & 15);
    float v = 0.f;
    if (!transpose) { if (K < rows && n < cols) v = src[K * cols + n]; }
    else            { if (K < cols && n < rows) v = src[n * cols + K]; }
    dst[idx] = (_Float16)v;
  }
}

__global__ void k_pack_weights(const float* W1, const float* W2, const float* W3,
                               const float* W4, const float* Wmix, _Float16* pack) {
  int L   = blockIdx.y;
  int tid = blockIdx.x * blockDim.x + threadIdx.x;
  int nth = gridDim.x * blockDim.x;
  _Float16* p = pack + (size_t)L * PACK_PER_LAYER;
  const float* w1 = W1 + L * 8 * 64;
  const float* w2 = W2 + L * 64 * 64;
  const float* w3 = W3 + L * 64 * 64;
  const float* w4 = W4 + L * 64 * 128;
  const float* wm = Wmix + L * 4 * 32 * 32;
  packB(w1, 8, 64, false, 1, 4, p + OFF_W1F, tid, nth);
  packB(w2, 64, 64, false, 2, 4, p + OFF_W2F, tid, nth);
  packB(w3, 64, 64, false, 2, 4, p + OFF_W3F, tid, nth);
  packB(w4, 64, 128, false, 2, 8, p + OFF_W4F, tid, nth);
  packB(w4, 64, 128, true, 4, 4, p + OFF_W4T, tid, nth);
  packB(w3, 64, 64, true, 2, 4, p + OFF_W3T, tid, nth);
  packB(w2, 64, 64, true, 2, 4, p + OFF_W2T, tid, nth);
  packB(w1, 8, 64, true, 2, 1, p + OFF_W1T, tid, nth);
  for (int l = 0; l < 4; ++l) {
    packB(wm + l * 1024, 32, 32, false, 1, 2, p + OFF_WMIX + l * 1024, tid, nth);
    packB(wm + l * 1024, 32, 32, true,  1, 2, p + OFF_WMIXT + l * 1024, tid, nth);
  }
}

// ---------------------------------------------------------------- geometry
__global__ void k_edge_geom(const float* pos, const float* cell, const float* shifts,
                            const int* ei, const int* ej, const int* batch,
                            float* RijB, float* rB, float* Ybuf, float* radB) {
  int e = blockIdx.x * blockDim.x + threadIdx.x;
  if (e >= KE) return;
  int i = ei[e], j = ej[e];
  const float* C = cell + batch[i] * 9;
  float sx = shifts[e*3+0], sy = shifts[e*3+1], sz = shifts[e*3+2];
  float shv0 = sx*C[0] + sy*C[3] + sz*C[6];
  float shv1 = sx*C[1] + sy*C[4] + sz*C[7];
  float shv2 = sx*C[2] + sy*C[5] + sz*C[8];
  float Rx = (pos[j*3+0] - pos[i*3+0] + shv0) * INV_CUT;
  float Ry = (pos[j*3+1] - pos[i*3+1] + shv1) * INV_CUT;
  float Rz = (pos[j*3+2] - pos[i*3+2] + shv2) * INV_CUT;
  float r  = sqrtf(Rx*Rx + Ry*Ry + Rz*Rz);
  float rc = fmaxf(r, 1e-9f);
  float x = Rx/rc, y = Ry/rc, z = Rz/rc;
  RijB[e*3+0] = Rx; RijB[e*3+1] = Ry; RijB[e*3+2] = Rz; rB[e] = r;
  const float s3=1.7320508075688772f, s15=3.872983346207417f, s5=2.23606797749979f;
  const float s70=8.366600265340756f, s105=10.246950765959598f, s42=6.48074069840786f, s7=2.6457513110645907f;
  float x2=x*x, y2=y*y, z2=z*z;
  float* Y = Ybuf + (size_t)e*16;
  Y[0]=1.f; Y[1]=s3*x; Y[2]=s3*y; Y[3]=s3*z;
  Y[4]=s15*x*y; Y[5]=s15*y*z; Y[6]=0.5f*s5*(3.f*z2-1.f); Y[7]=s15*x*z; Y[8]=0.5f*s15*(x2-y2);
  Y[9]=0.25f*s70*y*(3.f*x2-y2); Y[10]=s105*x*y*z; Y[11]=0.25f*s42*y*(5.f*z2-1.f);
  Y[12]=0.5f*s7*(5.f*z2*z-3.f*z); Y[13]=0.25f*s42*x*(5.f*z2-1.f);
  Y[14]=0.5f*s105*z*(x2-y2); Y[15]=0.25f*s70*x*(x2-3.f*y2);
  float r2=r*r, r4=r2*r2, r6=r4*r2, r7=r6*r, r8=r7*r;
  float fcut = (r < 1.f) ? (1.f - 28.f*r6 + 48.f*r7 - 21.f*r8) : 0.f;
  const float SQ2 = 1.4142135623730951f, PIf = 3.14159265358979f;
#pragma unroll
  for (int n = 1; n <= 8; ++n)
    radB[(size_t)e*8 + n - 1] = SQ2 * __sinf(PIf * n * r) / rc * fcut;
}

// ---------------------------------------------------------------- init / misc
__global__ void k_init_feats(const float* W_emb, float* feats, float* hj0) {
  int t = blockIdx.x * blockDim.x + threadIdx.x;
  if (t >= KN * KF * 16) return;
  int s = t & 15, c = (t >> 4) & 31;
  float v = (s == 0) ? W_emb[c] : 0.f;
  feats[t] = v;
  if (s == 0) hj0[t >> 4] = v;
}

__global__ void k_var_ones(float* var) { if (threadIdx.x < KG) var[threadIdx.x] = 1.0f; }

__global__ void k_readout(const float* hj, const float* w_read, int L, const int* batch, float* energy) {
  int n = blockIdx.x * blockDim.x + threadIdx.x;
  if (n >= KN) return;
  const float* w = w_read + L * KF;
  float acc = 0.f;
#pragma unroll
  for (int c = 0; c < KF; ++c) acc += hj[n * KF + c] * w[c];
  atomicAdd(&energy[batch[n]], acc);
}

// ---------------------------------------------------------------- forward edge MLP + message scatter
__global__ __launch_bounds__(32)
void k_edge_mlp_fwd(const float* radB, const float* Ybuf, const float* hjprev,
                    const int* ei, const int* ej, const _Float16* pk,
                    const float* b1, const float* b2, const float* b3, const float* b4,
                    float* agg) {
  __shared__ _Float16 tile[16 * 64];
  __shared__ float rwT[16 * 128];
  int lane = threadIdx.x, row = lane & 15, hi = lane >> 4;
  int e0 = blockIdx.x * 16;
  v8f acc[8];
  // stage 1: [16x8(pad32)] @ [32x64]
  v16h a;
#pragma unroll
  for (int e = 0; e < 16; ++e) {
    int K = e + ((e & 8) ? 8 : 0) + hi * 8;
    a[e] = (K < 8) ? (_Float16)radB[(size_t)(e0 + row) * 8 + K] : (_Float16)0.f;
  }
#pragma unroll
  for (int nt = 0; nt < 4; ++nt) {
    v8f c = wmma16(a, bfrag(pk + OFF_W1F, nt, lane), v8zero());
#pragma unroll
    for (int v = 0; v < 8; ++v)
      tile[(v + hi * 8) * 64 + nt * 16 + row] = (_Float16)siluf(c[v] + b1[nt * 16 + row]);
  }
  __syncthreads();
  // stage 2
  v16h a0 = afrag_h(tile, 64, 0, lane), a1 = afrag_h(tile, 64, 1, lane);
#pragma unroll
  for (int nt = 0; nt < 4; ++nt) {
    v8f c = wmma16(a0, bfrag(pk + OFF_W2F, 0 * 4 + nt, lane), v8zero());
    c = wmma16(a1, bfrag(pk + OFF_W2F, 1 * 4 + nt, lane), c);
    acc[nt] = c;
  }
  __syncthreads();
#pragma unroll
  for (int nt = 0; nt < 4; ++nt)
#pragma unroll
    for (int v = 0; v < 8; ++v)
      tile[(v + hi * 8) * 64 + nt * 16 + row] = (_Float16)siluf(acc[nt][v] + b2[nt * 16 + row]);
  __syncthreads();
  // stage 3
  a0 = afrag_h(tile, 64, 0, lane); a1 = afrag_h(tile, 64, 1, lane);
#pragma unroll
  for (int nt = 0; nt < 4; ++nt) {
    v8f c = wmma16(a0, bfrag(pk + OFF_W3F, 0 * 4 + nt, lane), v8zero());
    c = wmma16(a1, bfrag(pk + OFF_W3F, 1 * 4 + nt, lane), c);
    acc[nt] = c;
  }
  __syncthreads();
#pragma unroll
  for (int nt = 0; nt < 4; ++nt)
#pragma unroll
    for (int v = 0; v < 8; ++v)
      tile[(v + hi * 8) * 64 + nt * 16 + row] = (_Float16)siluf(acc[nt][v] + b3[nt * 16 + row]);
  __syncthreads();
  // stage 4: Rw = h3 @ W4 + b4  ->  rwT[16][128]
  a0 = afrag_h(tile, 64, 0, lane); a1 = afrag_h(tile, 64, 1, lane);
#pragma unroll
  for (int nt = 0; nt < 8; ++nt) {
    v8f c = wmma16(a0, bfrag(pk + OFF_W4F, 0 * 8 + nt, lane), v8zero());
    c = wmma16(a1, bfrag(pk + OFF_W4F, 1 * 8 + nt, lane), c);
#pragma unroll
    for (int v = 0; v < 8; ++v)
      rwT[(v + hi * 8) * 128 + nt * 16 + row] = c[v] + b4[nt * 16 + row];
  }
  __syncthreads();
  // message scatter: lane == channel f
  int f = lane;
  for (int e = 0; e < 16; ++e) {
    int edge = e0 + e;
    int i = ei[edge], j = ej[edge];
    float hjf = hjprev[j * KF + f];
    const float* Ye = Ybuf + (size_t)edge * 16;
    float rw[4];
#pragma unroll
    for (int kk = 0; kk < 4; ++kk) rw[kk] = rwT[e * 128 + f * 4 + kk];
    float* base = agg + ((size_t)i * KF + f) * 16;
#pragma unroll
    for (int s = 0; s < 16; ++s) {
      const int k = (s == 0) ? 0 : (s < 4) ? 1 : (s < 9) ? 2 : 3;
      atomicAdd(base + s, rw[k] * Ye[s] * hjf * AVG_INV);
    }
  }
}

// ---------------------------------------------------------------- channel mixing + gate + residual
__global__ __launch_bounds__(32)
void k_node_mix(const float* agg, const _Float16* wm, const float* pcoef, int L,
                float* feats, float* mixedOut, float* hjnext) {
  int lane = threadIdx.x, row = lane & 15, hi = lane >> 4;
  int n0 = blockIdx.x * 16;
  float p0 = pcoef[L * 2], p1 = pcoef[L * 2 + 1];
  float gate[2][8];
#pragma unroll 1
  for (int s = 0; s < 16; ++s) {
    int l = (s == 0) ? 0 : (s < 4) ? 1 : (s < 9) ? 2 : 3;
    v16h a;
#pragma unroll
    for (int e = 0; e < 16; ++e) {
      int K = e + ((e & 8) ? 8 : 0) + hi * 8;
      a[e] = (_Float16)agg[((size_t)(n0 + row) * KF + K) * 16 + s];
    }
    v8f c0 = wmma16(a, bfrag(wm, l * 2 + 0, lane), v8zero());
    v8f c1 = wmma16(a, bfrag(wm, l * 2 + 1, lane), v8zero());
#pragma unroll
    for (int nt = 0; nt < 2; ++nt)
#pragma unroll
      for (int v = 0; v < 8; ++v) {
        float m = nt ? c1[v] : c0[v];
        int n = n0 + v + hi * 8, cc = nt * 16 + row;
        size_t idx = ((size_t)n * KF + cc) * 16 + s;
        if (s == 0) gate[nt][v] = 1.f + p0 * m + p1 * m * m;
        mixedOut[idx] = m;
        float fv = m * gate[nt][v] + feats[idx];
        feats[idx] = fv;
        if (s == 0) hjnext[n * KF + cc] = fv;
      }
  }
}

// ---------------------------------------------------------------- backward: gate grad
__global__ void k_bwd_gate(float* gradF, const float* mixed, const float* w_read,
                           const float* pcoef, int L, float* dmix) {
  int t = blockIdx.x * blockDim.x + threadIdx.x;
  if (t >= KN * KF) return;
  int c = t & 31;
  float p0 = pcoef[L * 2], p1 = pcoef[L * 2 + 1];
  float* g = gradF + (size_t)t * 16;
  const float* m = mixed + (size_t)t * 16;
  float* dm = dmix + (size_t)t * 16;
  float g0 = g[0] + w_read[L * KF + c];
  g[0] = g0;
  float m0 = m[0];
  float gate = 1.f + p0 * m0 + p1 * m0 * m0;
  float dgate = p0 + 2.f * p1 * m0;
  float acc = 0.f;
#pragma unroll
  for (int s = 0; s < 16; ++s) {
    float gs = (s == 0) ? g0 : g[s];
    acc += gs * m[s];
    dm[s] = gs * gate;
  }
  dm[0] += acc * dgate;
}

// ---------------------------------------------------------------- backward: dagg = dmixed @ Wmix^T
__global__ __launch_bounds__(32)
void k_dagg(const float* dmix, const _Float16* wmT, float* dagg) {
  int lane = threadIdx.x, row = lane & 15, hi = lane >> 4;
  int n0 = blockIdx.x * 16;
#pragma unroll 1
  for (int s = 0; s < 16; ++s) {
    int l = (s == 0) ? 0 : (s < 4) ? 1 : (s < 9) ? 2 : 3;
    v16h a;
#pragma unroll
    for (int e = 0; e < 16; ++e) {
      int K = e + ((e & 8) ? 8 : 0) + hi * 8;
      a[e] = (_Float16)dmix[((size_t)(n0 + row) * KF + K) * 16 + s];
    }
    v8f c0 = wmma16(a, bfrag(wmT, l * 2 + 0, lane), v8zero());
    v8f c1 = wmma16(a, bfrag(wmT, l * 2 + 1, lane), v8zero());
#pragma unroll
    for (int nt = 0; nt < 2; ++nt)
#pragma unroll
      for (int v = 0; v < 8; ++v) {
        int n = n0 + v + hi * 8, d = nt * 16 + row;
        dagg[((size_t)n * KF + d) * 16 + s] = nt ? c1[v] : c0[v];
      }
  }
}

// ---------------------------------------------------------------- backward: edges (recompute + grads)
__global__ __launch_bounds__(32)
void k_edge_bwd(const float* radB, const float* Ybuf, const float* hjprev,
                const int* ei, const int* ej, const _Float16* pk,
                const float* b1, const float* b2, const float* b3, const float* b4,
                const float* dagg, float* gradF, float* dYb, float* dradB) {
  __shared__ _Float16 tile[16 * 64];
  __shared__ float rwT[16 * 128];
  __shared__ float drwT[16 * 128];
  __shared__ float dYl[256];
  int lane = threadIdx.x, row = lane & 15, hi = lane >> 4;
  int e0 = blockIdx.x * 16;
  v8f z1[4], z2[4], z3[4];
  // ---- forward recompute (keep pre-activations) ----
  v16h a;
#pragma unroll
  for (int e = 0; e < 16; ++e) {
    int K = e + ((e & 8) ? 8 : 0) + hi * 8;
    a[e] = (K < 8) ? (_Float16)radB[(size_t)(e0 + row) * 8 + K] : (_Float16)0.f;
  }
#pragma unroll
  for (int nt = 0; nt < 4; ++nt) {
    v8f c = wmma16(a, bfrag(pk + OFF_W1F, nt, lane), v8zero());
#pragma unroll
    for (int v = 0; v < 8; ++v) c[v] += b1[nt * 16 + row];
    z1[nt] = c;
#pragma unroll
    for (int v = 0; v < 8; ++v)
      tile[(v + hi * 8) * 64 + nt * 16 + row] = (_Float16)siluf(c[v]);
  }
  __syncthreads();
  v16h a0 = afrag_h(tile, 64, 0, lane), a1 = afrag_h(tile, 64, 1, lane);
  __syncthreads();
#pragma unroll
  for (int nt = 0; nt < 4; ++nt) {
    v8f c = wmma16(a0, bfrag(pk + OFF_W2F, nt, lane), v8zero());
    c = wmma16(a1, bfrag(pk + OFF_W2F, 4 + nt, lane), c);
#pragma unroll
    for (int v = 0; v < 8; ++v) c[v] += b2[nt * 16 + row];
    z2[nt] = c;
#pragma unroll
    for (int v = 0; v < 8; ++v)
      tile[(v + hi * 8) * 64 + nt * 16 + row] = (_Float16)siluf(c[v]);
  }
  __syncthreads();
  a0 = afrag_h(tile, 64, 0, lane); a1 = afrag_h(tile, 64, 1, lane);
  __syncthreads();
#pragma unroll
  for (int nt = 0; nt < 4; ++nt) {
    v8f c = wmma16(a0, bfrag(pk + OFF_W3F, nt, lane), v8zero());
    c = wmma16(a1, bfrag(pk + OFF_W3F, 4 + nt, lane), c);
#pragma unroll
    for (int v = 0; v < 8; ++v) c[v] += b3[nt * 16 + row];
    z3[nt] = c;
#pragma unroll
    for (int v = 0; v < 8; ++v)
      tile[(v + hi * 8) * 64 + nt * 16 + row] = (_Float16)siluf(c[v]);
  }
  __syncthreads();
  a0 = afrag_h(tile, 64, 0, lane); a1 = afrag_h(tile, 64, 1, lane);
#pragma unroll
  for (int nt = 0; nt < 8; ++nt) {
    v8f c = wmma16(a0, bfrag(pk + OFF_W4F, nt, lane), v8zero());
    c = wmma16(a1, bfrag(pk + OFF_W4F, 8 + nt, lane), c);
#pragma unroll
    for (int v = 0; v < 8; ++v)
      rwT[(v + hi * 8) * 128 + nt * 16 + row] = c[v] + b4[nt * 16 + row];
  }
  for (int t = lane; t < 256; t += 32) dYl[t] = 0.f;
  __syncthreads();
  // ---- per-edge reductions (lane == channel f) ----
  int f = lane;
  for (int e = 0; e < 16; ++e) {
    int edge = e0 + e;
    int i = ei[edge], j = ej[edge];
    float hjf = hjprev[j * KF + f];
    const float* da = dagg + ((size_t)i * KF + f) * 16;
    const float* Ye = Ybuf + (size_t)edge * 16;
    float rw[4];
#pragma unroll
    for (int kk = 0; kk < 4; ++kk) rw[kk] = rwT[e * 128 + f * 4 + kk];
    float drw[4] = {0.f, 0.f, 0.f, 0.f};
    float dhj = 0.f;
#pragma unroll
    for (int s = 0; s < 16; ++s) {
      const int k = (s == 0) ? 0 : (s < 4) ? 1 : (s < 9) ? 2 : 3;
      float dm = da[s] * AVG_INV;
      float ys = Ye[s];
      dhj += dm * rw[k] * ys;
      drw[k] += dm * ys;
      atomicAdd(&dYl[e * 16 + s], dm * rw[k] * hjf);
    }
    atomicAdd(&gradF[((size_t)j * KF + f) * 16], dhj);
#pragma unroll
    for (int kk = 0; kk < 4; ++kk) drwT[e * 128 + f * 4 + kk] = drw[kk] * hjf;
  }
  __syncthreads();
  for (int t = lane; t < 256; t += 32) {
    int e = t >> 4, s = t & 15;
    dYb[(size_t)(e0 + e) * 16 + s] += dYl[t];
  }
  // ---- MLP backward ----
  v8f d3[4];
#pragma unroll
  for (int nt = 0; nt < 4; ++nt) {
    v8f c = v8zero();
#pragma unroll
    for (int kt = 0; kt < 4; ++kt)
      c = wmma16(afrag_f(drwT, 128, kt, lane), bfrag(pk + OFF_W4T, kt * 4 + nt, lane), c);
    d3[nt] = c;
  }
  __syncthreads();
#pragma unroll
  for (int nt = 0; nt < 4; ++nt)
#pragma unroll
    for (int v = 0; v < 8; ++v)
      tile[(v + hi * 8) * 64 + nt * 16 + row] = (_Float16)(d3[nt][v] * dsiluf(z3[nt][v]));
  __syncthreads();
  a0 = afrag_h(tile, 64, 0, lane); a1 = afrag_h(tile, 64, 1, lane);
#pragma unroll
  for (int nt = 0; nt < 4; ++nt) {
    v8f c = wmma16(a0, bfrag(pk + OFF_W3T, nt, lane), v8zero());
    c = wmma16(a1, bfrag(pk + OFF_W3T, 4 + nt, lane), c);
    d3[nt] = c;
  }
  __syncthreads();
#pragma unroll
  for (int nt = 0; nt < 4; ++nt)
#pragma unroll
    for (int v = 0; v < 8; ++v)
      tile[(v + hi * 8) * 64 + nt * 16 + row] = (_Float16)(d3[nt][v] * dsiluf(z2[nt][v]));
  __syncthreads();
  a0 = afrag_h(tile, 64, 0, lane); a1 = afrag_h(tile, 64, 1, lane);
#pragma unroll
  for (int nt = 0; nt < 4; ++nt) {
    v8f c = wmma16(a0, bfrag(pk + OFF_W2T, nt, lane), v8zero());
    c = wmma16(a1, bfrag(pk + OFF_W2T, 4 + nt, lane), c);
    d3[nt] = c;
  }
  __syncthreads();
#pragma unroll
  for (int nt = 0; nt < 4; ++nt)
#pragma unroll
    for (int v = 0; v < 8; ++v)
      tile[(v + hi * 8) * 64 + nt * 16 + row] = (_Float16)(d3[nt][v] * dsiluf(z1[nt][v]));
  __syncthreads();
  a0 = afrag_h(tile, 64, 0, lane); a1 = afrag_h(tile, 64, 1, lane);
  {
    v8f c = wmma16(a0, bfrag(pk + OFF_W1T, 0, lane), v8zero());
    c = wmma16(a1, bfrag(pk + OFF_W1T, 1, lane), c);
    int b = lane & 15;
    if (b < 8) {
#pragma unroll
      for (int v = 0; v < 8; ++v) {
        int edge = e0 + v + hi * 8;
        dradB[(size_t)edge * 8 + b] += c[v];
      }
    }
  }
}

// ---------------------------------------------------------------- forces
__global__ void k_edge_force(const float* RijB, const float* rB, const float* dYb,
                             const float* dradB, const int* ei, const int* ej, float* forces) {
  int e = blockIdx.x * blockDim.x + threadIdx.x;
  if (e >= KE) return;
  float r = rB[e];
  float rc = fmaxf(r, 1e-9f);
  float x = RijB[e*3+0]/rc, y = RijB[e*3+1]/rc, z = RijB[e*3+2]/rc;
  // radial part
  float r2=r*r, r4=r2*r2, r5=r4*r, r6=r5*r, r7=r6*r;
  float fcut = 0.f, dfcut = 0.f;
  if (r < 1.f) {
    fcut  = 1.f - 28.f*r6 + 48.f*r7 - 21.f*r7*r;
    dfcut = -168.f*r5 + 336.f*r6 - 168.f*r7;
  }
  const float SQ2 = 1.4142135623730951f, PIf = 3.14159265358979f;
  float gr = 0.f;
#pragma unroll
  for (int n = 1; n <= 8; ++n) {
    float arg = PIf * n * r;
    float sn = __sinf(arg), cn = __cosf(arg);
    float bess  = SQ2 * sn / rc;
    float dbess = SQ2 * (PIf * n * cn * rc - sn) / (rc * rc);
    gr += dradB[(size_t)e*8 + n - 1] * (dbess * fcut + bess * dfcut);
  }
  // angular part
  const float s3=1.7320508075688772f, s15=3.872983346207417f, s5=2.23606797749979f;
  const float s70=8.366600265340756f, s105=10.246950765959598f, s42=6.48074069840786f, s7=2.6457513110645907f;
  const float* dY = dYb + (size_t)e*16;
  float x2=x*x, y2=y*y, z2=z*z;
  float gx=0.f, gy=0.f, gz=0.f;
  gx += dY[1]*s3;                         gy += dY[2]*s3;                        gz += dY[3]*s3;
  gx += dY[4]*s15*y;                      gy += dY[4]*s15*x;
  gy += dY[5]*s15*z;                      gz += dY[5]*s15*y;
  gz += dY[6]*3.f*s5*z;
  gx += dY[7]*s15*z;                      gz += dY[7]*s15*x;
  gx += dY[8]*s15*x;                      gy -= dY[8]*s15*y;
  gx += dY[9]*1.5f*s70*x*y;               gy += dY[9]*0.75f*s70*(x2-y2);
  gx += dY[10]*s105*y*z;                  gy += dY[10]*s105*x*z;                 gz += dY[10]*s105*x*y;
  gy += dY[11]*0.25f*s42*(5.f*z2-1.f);    gz += dY[11]*2.5f*s42*y*z;
  gz += dY[12]*0.5f*s7*(15.f*z2-3.f);
  gx += dY[13]*0.25f*s42*(5.f*z2-1.f);    gz += dY[13]*2.5f*s42*x*z;
  gx += dY[14]*s105*x*z;                  gy -= dY[14]*s105*y*z;                 gz += dY[14]*0.5f*s105*(x2-y2);
  gx += dY[15]*0.75f*s70*(x2-y2);         gy -= dY[15]*1.5f*s70*x*y;
  float gdu = gx*x + gy*y + gz*z;
  float inv_r = 1.f / rc;
  float dEx = gr*x + (gx - gdu*x) * inv_r;
  float dEy = gr*y + (gy - gdu*y) * inv_r;
  float dEz = gr*z + (gz - gdu*z) * inv_r;
  int i = ei[e], j = ej[e];
  // forces = -dE/dpos ; dpos_j = +dEdRij/CUT, dpos_i = -dEdRij/CUT
  atomicAdd(&forces[j*3+0], -dEx * INV_CUT);
  atomicAdd(&forces[j*3+1], -dEy * INV_CUT);
  atomicAdd(&forces[j*3+2], -dEz * INV_CUT);
  atomicAdd(&forces[i*3+0],  dEx * INV_CUT);
  atomicAdd(&forces[i*3+1],  dEy * INV_CUT);
  atomicAdd(&forces[i*3+2],  dEz * INV_CUT);
}

// ---------------------------------------------------------------- host
extern "C" void kernel_launch(void* const* d_in, const int* in_sizes, int n_in,
                              void* d_out, int out_size, void* d_ws, size_t ws_size,
                              hipStream_t stream) {
  (void)in_sizes; (void)n_in; (void)ws_size;
  const float* pos    = (const float*)d_in[0];
  const float* cell   = (const float*)d_in[1];
  const float* shifts = (const float*)d_in[2];
  const float* W_emb  = (const float*)d_in[3];
  const float* W1 = (const float*)d_in[4];
  const float* b1 = (const float*)d_in[5];
  const float* W2 = (const float*)d_in[6];
  const float* b2 = (const float*)d_in[7];
  const float* W3 = (const float*)d_in[8];
  const float* b3 = (const float*)d_in[9];
  const float* W4 = (const float*)d_in[10];
  const float* b4 = (const float*)d_in[11];
  const float* Wmix   = (const float*)d_in[12];
  const float* w_read = (const float*)d_in[13];
  const float* pcoef  = (const float*)d_in[14];
  const int* eidx  = (const int*)d_in[15];
  const int* batch = (const int*)d_in[16];
  const int* ei = eidx;
  const int* ej = eidx + KE;
  float* energy = (float*)d_out;
  float* forces = (float*)d_out + 32;

  char* w = (char*)d_ws;
  auto balloc = [&](size_t bytes) -> char* {
    char* p = w;
    w += (bytes + 255) & ~(size_t)255;
    return p;
  };
  float* Ybuf   = (float*)balloc((size_t)KE * 16 * 4);
  float* radB   = (float*)balloc((size_t)KE * 8 * 4);
  float* RijB   = (float*)balloc((size_t)KE * 3 * 4);
  float* rB     = (float*)balloc((size_t)KE * 4);
  float* feats  = (float*)balloc((size_t)KN * KF * 16 * 4);
  float* hjb    = (float*)balloc((size_t)4 * KN * KF * 4);
  float* mixedB = (float*)balloc((size_t)3 * KN * KF * 16 * 4);
  float* aggB   = (float*)balloc((size_t)KN * KF * 16 * 4);
  float* dmixB  = (float*)balloc((size_t)KN * KF * 16 * 4);
  float* daggB  = (float*)balloc((size_t)KN * KF * 16 * 4);
  float* gradF  = (float*)balloc((size_t)KN * KF * 16 * 4);
  float* dYb    = (float*)balloc((size_t)KE * 16 * 4);
  float* dradB  = (float*)balloc((size_t)KE * 8 * 4);
  _Float16* pack = (_Float16*)balloc((size_t)3 * PACK_PER_LAYER * sizeof(_Float16));

  hipMemsetAsync(d_out, 0, (size_t)out_size * sizeof(float), stream);
  k_var_ones<<<1, 32, 0, stream>>>(energy + 16);
  k_pack_weights<<<dim3(64, 3), 256, 0, stream>>>(W1, W2, W3, W4, Wmix, pack);
  k_edge_geom<<<(KE + 255) / 256, 256, 0, stream>>>(pos, cell, shifts, ei, ej, batch,
                                                    RijB, rB, Ybuf, radB);
  k_init_feats<<<(KN * KF * 16 + 255) / 256, 256, 0, stream>>>(W_emb, feats, hjb);

  for (int L = 0; L < 3; ++L) {
    const _Float16* pk = pack + (size_t)L * PACK_PER_LAYER;
    hipMemsetAsync(aggB, 0, (size_t)KN * KF * 16 * 4, stream);
    k_edge_mlp_fwd<<<KE / 16, 32, 0, stream>>>(radB, Ybuf, hjb + (size_t)L * KN * KF,
                                               ei, ej, pk,
                                               b1 + L * 64, b2 + L * 64, b3 + L * 64, b4 + L * 128,
                                               aggB);
    k_node_mix<<<KN / 16, 32, 0, stream>>>(aggB, pk + OFF_WMIX, pcoef, L,
                                           feats, mixedB + (size_t)L * KN * KF * 16,
                                           hjb + (size_t)(L + 1) * KN * KF);
    k_readout<<<(KN + 255) / 256, 256, 0, stream>>>(hjb + (size_t)(L + 1) * KN * KF,
                                                    w_read, L, batch, energy);
  }

  hipMemsetAsync(gradF, 0, (size_t)KN * KF * 16 * 4, stream);
  hipMemsetAsync(dYb,   0, (size_t)KE * 16 * 4, stream);
  hipMemsetAsync(dradB, 0, (size_t)KE * 8 * 4, stream);

  for (int L = 2; L >= 0; --L) {
    const _Float16* pk = pack + (size_t)L * PACK_PER_LAYER;
    k_bwd_gate<<<(KN * KF + 255) / 256, 256, 0, stream>>>(gradF,
                                                          mixedB + (size_t)L * KN * KF * 16,
                                                          w_read, pcoef, L, dmixB);
    k_dagg<<<KN / 16, 32, 0, stream>>>(dmixB, pk + OFF_WMIXT, daggB);
    k_edge_bwd<<<KE / 16, 32, 0, stream>>>(radB, Ybuf, hjb + (size_t)L * KN * KF,
                                           ei, ej, pk,
                                           b1 + L * 64, b2 + L * 64, b3 + L * 64, b4 + L * 128,
                                           daggB, gradF, dYb, dradB);
  }

  k_edge_force<<<(KE + 255) / 256, 256, 0, stream>>>(RijB, rB, dYb, dradB, ei, ej, forces);
}
